// Poly2d_83751862272438
// MI455X (gfx1250) — matrix operand order repaired and checked by reading, not compiled, not run on hardware
//
#include <hip/hip_runtime.h>

typedef _Float16 half_t;
typedef __attribute__((ext_vector_type(8)))  _Float16 v8h;
typedef __attribute__((ext_vector_type(16))) _Float16 v16h;
typedef __attribute__((ext_vector_type(8)))  float    v8f;
typedef __attribute__((ext_vector_type(4)))  float    v4f;

// ---------------------------------------------------------------------------
// Problem constants (match reference: bs=8, p_in=64, H=W=64, p_out=64, vs=10)
// ---------------------------------------------------------------------------
#define BS     8
#define CIN    64
#define HH     64
#define WW     64
#define COUT   64
#define MTOT   (BS * HH * WW)      // 32768 pixels
#define MBLK   128                 // M tile per workgroup (2 image rows)
#define QPAD   64                  // 55 symmetric pairs padded to 64 (2 K-chunks of 32)
#define KTOT   (CIN * QPAD)        // 4096
#define BP_ELEMS (KTOT * COUT)     // 262144 f16 in workspace (512 KB)

#define AROW   40                  // halves per A-tile row (32 data + 8 pad = 80 B)
#define ATILE  (MBLK * AROW)       // halves per A buffer
#define XROW   66                  // floats per x-tile row (64 + 2 halo)
#define XTILE  (4 * XROW)          // floats per x-tile buffer (264)

// Upper-triangular pair enumeration (i<=j) of the 10-vector s = [1, patch9].
constexpr int PAIR_I[55] = {
  0,0,0,0,0,0,0,0,0,0,
  1,1,1,1,1,1,1,1,1,
  2,2,2,2,2,2,2,2,
  3,3,3,3,3,3,3,
  4,4,4,4,4,4,
  5,5,5,5,5,
  6,6,6,6,
  7,7,7,
  8,8,
  9};
constexpr int PAIR_J[55] = {
  0,1,2,3,4,5,6,7,8,9,
  1,2,3,4,5,6,7,8,9,
  2,3,4,5,6,7,8,9,
  3,4,5,6,7,8,9,
  4,5,6,7,8,9,
  5,6,7,8,9,
  6,7,8,9,
  7,8,9,
  8,9,
  9};

template <int Q>
__device__ __forceinline__ half_t pairprod(const half_t* hs) {
  if constexpr (Q < 55) {
    return hs[PAIR_I[Q]] * hs[PAIR_J[Q]];   // compile-time indices; f16 muls
  } else {
    return (half_t)0.0f;                    // zero padding q = 55..63
  }
}

template <int QB>
__device__ __forceinline__ v8h build8(const half_t* hs) {
  v8h p;
  p[0] = pairprod<QB + 0>(hs);
  p[1] = pairprod<QB + 1>(hs);
  p[2] = pairprod<QB + 2>(hs);
  p[3] = pairprod<QB + 3>(hs);
  p[4] = pairprod<QB + 4>(hs);
  p[5] = pairprod<QB + 5>(hs);
  p[6] = pairprod<QB + 6>(hs);
  p[7] = pairprod<QB + 7>(hs);
  return p;
}

// ---------------------------------------------------------------------------
// Prep: fold filters into symmetric-pair coefficients, convert to f16, pack
// into WMMA-B fragment order (frag f = kc*4+nt; lane's 16 halves contiguous).
// ---------------------------------------------------------------------------
__global__ __launch_bounds__(256) void poly2d_prep(
    const float* __restrict__ filters, half_t* __restrict__ Bp) {
  int e = blockIdx.x * 256 + threadIdx.x;
  if (e >= BP_ELEMS) return;
  int elem  = e & 15;
  int lane  = (e >> 4) & 31;
  int f     = e >> 9;
  int kc    = f >> 2;
  int nt    = f & 3;
  int hsel  = lane >> 4;
  int n     = nt * 16 + (lane & 15);
  int K     = (elem < 8) ? hsel * 8 + elem : 16 + hsel * 8 + (elem - 8);
  int kglob = kc * 32 + K;
  int c     = kglob >> 6;
  int q     = kglob & 63;
  float wv = 0.0f;
  if (q < 55) {
    int i = 0, qq = q;
    while (qq >= 10 - i) { qq -= (10 - i); ++i; }
    int j = i + qq;
    const float* Fb = filters + ((size_t)n * CIN + c) * 100;  // F[n][c][10][10]
    wv = Fb[i * 10 + j];
    if (i != j) wv += Fb[j * 10 + i];  // symmetric fold
  }
  Bp[e] = (half_t)wv;
}

// ---------------------------------------------------------------------------
// One K-chunk MMA step for one wave.
// ---------------------------------------------------------------------------
__device__ __forceinline__ void mma_step(
    const half_t* __restrict__ arow, const half_t* __restrict__ Bp, int kc,
    int lane, int hsel, v8f& c0, v8f& c1, v8f& c2, v8f& c3) {
  v8h lo = *(const v8h*)(arow + 8 * hsel);        // K = 8h .. 8h+7
  v8h hi = *(const v8h*)(arow + 16 + 8 * hsel);   // K = 16+8h .. 23+8h
  v16h a;
#pragma unroll
  for (int i = 0; i < 8; ++i) { a[i] = lo[i]; a[i + 8] = hi[i]; }
  const half_t* bb = Bp + (size_t)kc * 2048 + lane * 16;  // 4 frags * 512 halves
  v16h b0 = *(const v16h*)(bb);
  v16h b1 = *(const v16h*)(bb + 512);
  v16h b2 = *(const v16h*)(bb + 1024);
  v16h b3 = *(const v16h*)(bb + 1536);
  c0 = __builtin_amdgcn_wmma_f32_16x16x32_f16(false, a, false, b0, (short)0, c0, false, false);
  c1 = __builtin_amdgcn_wmma_f32_16x16x32_f16(false, a, false, b1, (short)0, c1, false, false);
  c2 = __builtin_amdgcn_wmma_f32_16x16x32_f16(false, a, false, b2, (short)0, c2, false, false);
  c3 = __builtin_amdgcn_wmma_f32_16x16x32_f16(false, a, false, b3, (short)0, c3, false, false);
}

// ---------------------------------------------------------------------------
// Main kernel: 256 threads = 8 waves; output tile 128(M) x 64(N).
// Channel pipeline (2 barriers / channel, double-buffered A tile and x tile):
//   phase A: async-DMA x row-tile for channel c+1 into x-buf[c+1 & 1],
//            gather s from x-buf[c & 1] (no predication: halo pre-zeroed),
//            build K-chunk0 products -> A-bufA;  barrier;  mma(chunk0)
//   phase B: build K-chunk1 -> A-bufB;  s_wait_asynccnt 0;  barrier; mma(chunk1)
// Builder K-subhalf is wave-uniform (sub = t>>7, readfirstlane) so the build
// path selection is a scalar branch, not exec-masked dual execution.
// ---------------------------------------------------------------------------
__global__ __launch_bounds__(256) void poly2d_main(
    const float* __restrict__ x, const half_t* __restrict__ Bp,
    const float* __restrict__ bias, float* __restrict__ out) {
  __shared__ alignas(16) half_t smemA[2 * ATILE];
  __shared__ alignas(16) float  xtile[2 * XTILE];

  const int t      = threadIdx.x;
  const int Mblock = blockIdx.x * MBLK;
  const int bi     = Mblock >> 12;           // image index (block never spans images)
  const int h0     = (Mblock & 4095) >> 6;   // first image row of this block
  const float* ximg = x + (size_t)bi * CIN * (HH * WW);

  // ---- x-tile staging identity: one element per thread per channel ----
  const int xr   = t >> 6;                   // tile row 0..3  (image row h0-1+xr)
  const int xcol = t & 63;
  const int grow = h0 - 1 + xr;
  const bool rowok = (grow >= 0) && (grow < HH);
  const int growc  = rowok ? grow : 0;
  const int goff   = (growc * WW + xcol) * 4;            // byte offset within channel plane
  const int xti    = xr * XROW + 1 + xcol;               // element within x-tile buffer

  // ---- builder identity: row mb (0..127), wave-uniform K-subhalf sub ----
  const int mb  = t & 127;
  const int sub = __builtin_amdgcn_readfirstlane(t >> 7);   // uniform per wave
  const int hl  = mb >> 6;                   // 0/1: row within block
  const int wl  = mb & 63;
  half_t* dstA = smemA + mb * AROW + sub * 16;
  half_t* dstB = dstA + ATILE;

  // ---- wmma identity ----
  const int wave  = t >> 5;
  const int lane  = t & 31;
  const int hsel  = lane >> 4;
  const int nfrag = lane & 15;
  const half_t* arowA = smemA + (wave * 16 + nfrag) * AROW;
  const half_t* arowB = arowA + ATILE;

  v8f acc0 = {}, acc1 = {}, acc2 = {}, acc3 = {};

  // ---- prologue: zero x-tiles (halo + invalid rows stay zero forever),
  //      then async-DMA channel 0 into x-buf 0 ----
  for (int k = t; k < 2 * XTILE; k += 256) xtile[k] = 0.0f;
  __syncthreads();
  if (rowok) {
    unsigned ldsa = (unsigned)(size_t)(const void*)&xtile[xti];
    asm volatile("global_load_async_to_lds_b32 %0, %1, %2"
                 :: "v"(ldsa), "v"(goff), "s"(ximg) : "memory");
  }
  asm volatile("s_wait_asynccnt 0x0" ::: "memory");
  __syncthreads();

  half_t hs[10];
  hs[0] = (half_t)1.0f;

  for (int c = 0; c < CIN; ++c) {
    const int bufx = c & 1;

    // prefetch channel c+1 into the other x buffer (overlaps this channel's MMAs)
    if ((c < CIN - 1) && rowok) {
      unsigned ldsa = (unsigned)(size_t)(const void*)&xtile[(bufx ^ 1) * XTILE + xti];
      asm volatile("global_load_async_to_lds_b32 %0, %1, %2"
                   :: "v"(ldsa), "v"(goff), "s"(ximg + (size_t)(c + 1) * (HH * WW))
                   : "memory");
    }

    // gather this thread's augmented patch vector from LDS, convert to f16
    const float* xt = xtile + bufx * XTILE + (hl + 1) * XROW + 1 + wl;
#pragma unroll
    for (int dy = -1; dy <= 1; ++dy)
#pragma unroll
      for (int dx = -1; dx <= 1; ++dx)
        hs[1 + (dy + 1) * 3 + (dx + 1)] = (half_t)xt[dy * XROW + dx];

    // ---- K-chunk 0 (q = 0..31) into A-buf A ----
    if (sub == 0) { *(v8h*)dstA = build8<0>(hs);  *(v8h*)(dstA + 8) = build8<8>(hs); }
    else          { *(v8h*)dstA = build8<16>(hs); *(v8h*)(dstA + 8) = build8<24>(hs); }
    __syncthreads();
    mma_step(arowA, Bp, c * 2 + 0, lane, hsel, acc0, acc1, acc2, acc3);

    // ---- K-chunk 1 (q = 32..63, 55..63 zero) into A-buf B ----
    if (sub == 0) { *(v8h*)dstB = build8<32>(hs); *(v8h*)(dstB + 8) = build8<40>(hs); }
    else          { *(v8h*)dstB = build8<48>(hs); *(v8h*)(dstB + 8) = build8<56>(hs); }
    asm volatile("s_wait_asynccnt 0x0" ::: "memory");   // c+1 x-tile landed
    __syncthreads();   // also guarantees all waves done reading A-buf A
    mma_step(arowB, Bp, c * 2 + 1, lane, hsel, acc0, acc1, acc2, acc3);
  }

  // ---- epilogue: C/D layout lane holds N = lane&15, VGPR v -> M = 8*hsel + v.
  // NCHW output: the 8 rows per lane are consecutive pixels -> 2 b128 stores.
  const int mloc = (Mblock & 4095) + wave * 16 + hsel * 8;
  float* ob = out + (size_t)bi * COUT * 4096 + mloc;
#pragma unroll
  for (int nt = 0; nt < 4; ++nt) {
    const v8f& a = (nt == 0) ? acc0 : (nt == 1) ? acc1 : (nt == 2) ? acc2 : acc3;
    const int n = nt * 16 + nfrag;
    const float bv = bias[n];
    float* op = ob + (size_t)n * 4096;
    v4f r0 = { a[0] + bv, a[1] + bv, a[2] + bv, a[3] + bv };
    v4f r1 = { a[4] + bv, a[5] + bv, a[6] + bv, a[7] + bv };
    *(v4f*)op = r0;
    *(v4f*)(op + 4) = r1;
  }
}

// ---------------------------------------------------------------------------
extern "C" void kernel_launch(void* const* d_in, const int* in_sizes, int n_in,
                              void* d_out, int out_size, void* d_ws, size_t ws_size,
                              hipStream_t stream) {
  const float* x       = (const float*)d_in[0];  // (8, 64, 64, 64)
  const float* filters = (const float*)d_in[1];  // (64, 64, 10, 10)
  const float* biases  = (const float*)d_in[2];  // (64,)
  float* out = (float*)d_out;                    // (8, 64, 64, 64)
  half_t* Bp = (half_t*)d_ws;                    // 262144 f16 = 512 KB

  poly2d_prep<<<(BP_ELEMS + 255) / 256, 256, 0, stream>>>(filters, Bp);
  poly2d_main<<<MTOT / MBLK, 256, 0, stream>>>(x, Bp, biases, out);
}
